// PointNet2_28982439313471
// MI455X (gfx1250) — matrix operand order repaired and checked by reading, not compiled, not run on hardware
//
#include <hip/hip_runtime.h>

// ---------------------------------------------------------------------------
// PointNet++ reference: the returned value is ONLY fp_layer(coords, l1c, data,
// fp1_params). l1c = coords[fps(coords,1024)]. Everything else (SA1/SA2 MLPs,
// ball query, FP2) is dead code. Pipeline:
//   1) fps_centers_kernel : per-batch FPS -> centers (B,1024,3)
//      coords register-resident; argmax via wave32 shfl butterfly on packed
//      64-bit keys + one ds_max_u64 per wave; 2 barriers per step.
//   2) knn3_interp_kernel : 3-NN inverse-distance interp of centers (B,N,3)
//   3) fp1_mlp_kernel     : [data(6),interp(3)] -> 128 -> 128 -> 128 (BN+ReLU)
//                           via v_wmma_f32_16x16x32_f16, f32 accumulate.
// ---------------------------------------------------------------------------

#define B_    8
#define N_    8192
#define DDIM  6
#define S1    1024
#define CH    128
#define BN_INV_F 0.99999500003749968f   // 1/sqrt(1+1e-5)

typedef __attribute__((ext_vector_type(16))) _Float16 v16h;
typedef __attribute__((ext_vector_type(8)))  float    v8f;

union FragAB { v16h v; unsigned u[8]; };

// ---------------------------------------------------------------------------
// Kernel 1: farthest point sampling, one workgroup per batch, 1024 threads.
// Per-thread registers hold 8 points + 8 running distances. Key packing:
// high 32 = f32 bits of dist (monotone, dist >= 0), low 32 = ~index so that
// max-reduction picks the smallest index on ties (argmax first-occurrence).
// ---------------------------------------------------------------------------
__global__ __launch_bounds__(1024) void fps_centers_kernel(
    const float* __restrict__ coords, float* __restrict__ centers)
{
    __shared__ unsigned long long sKey[2];   // double-buffered winner key
    __shared__ float sC[2][3];               // double-buffered centroid
    const int b = blockIdx.x, tid = threadIdx.x;
    const float* cb = coords + (size_t)b * N_ * 3;

    constexpr int Q = N_ / 1024;             // 8 points per thread
    float px[Q], py[Q], pz[Q], dist[Q];
#pragma unroll
    for (int q = 0; q < Q; ++q) {
        const int i = q * 1024 + tid;
        px[q] = cb[i * 3 + 0];
        py[q] = cb[i * 3 + 1];
        pz[q] = cb[i * 3 + 2];
        dist[q] = 3.4e38f;
    }
    if (tid == 0) {
        sKey[0] = 0ull; sKey[1] = 0ull;
        sC[0][0] = cb[0]; sC[0][1] = cb[1]; sC[0][2] = cb[2];   // farthest=0
    }
    __syncthreads();

    for (int t = 0; t < S1; ++t) {
        const int cur = t & 1, nxt = cur ^ 1;
        const float cx = sC[cur][0], cy = sC[cur][1], cz = sC[cur][2];
        if (tid == 0) {
            float* c = centers + ((size_t)b * S1 + t) * 3;
            c[0] = cx; c[1] = cy; c[2] = cz;
        }
        unsigned long long key = 0ull;
#pragma unroll
        for (int q = 0; q < Q; ++q) {
            const float dx = px[q] - cx, dy = py[q] - cy, dz = pz[q] - cz;
            const float d  = dx * dx + dy * dy + dz * dz;
            const float nd = fminf(dist[q], d);
            dist[q] = nd;
            const int i = q * 1024 + tid;
            const unsigned long long k =
                ((unsigned long long)__float_as_uint(nd) << 32) |
                (unsigned long long)(0xFFFFFFFFu - (unsigned)i);
            key = k > key ? k : key;
        }
        // wave32 butterfly max (no barriers)
#pragma unroll
        for (int m = 16; m > 0; m >>= 1) {
            const unsigned long long o = __shfl_xor(key, m, 32);
            key = o > key ? o : key;
        }
        if ((tid & 31) == 0) atomicMax(&sKey[cur], key);   // ds_max_u64
        __syncthreads();

        const unsigned long long win = sKey[cur];
        const int far = (int)(0xFFFFFFFFu - (unsigned)(win & 0xFFFFFFFFull));
        if ((far & 1023) == tid) {            // owner publishes next centroid
            const int q = far >> 10;
            float ox = px[0], oy = py[0], oz = pz[0];
#pragma unroll
            for (int qq = 1; qq < Q; ++qq) {
                if (q == qq) { ox = px[qq]; oy = py[qq]; oz = pz[qq]; }
            }
            sC[nxt][0] = ox; sC[nxt][1] = oy; sC[nxt][2] = oz;
        }
        if (tid == 0) sKey[nxt] = 0ull;       // arm next iteration's slot
        __syncthreads();
    }
}

// ---------------------------------------------------------------------------
// Kernel 2: per point, 3 nearest centers -> inverse-distance interp of the
// center COORDINATES (that's what the reference does). Centers cached in LDS.
// ---------------------------------------------------------------------------
__global__ __launch_bounds__(256) void knn3_interp_kernel(
    const float* __restrict__ coords, const float* __restrict__ centers,
    float* __restrict__ interp)
{
    __shared__ float sx[S1], sy[S1], sz[S1];
    const int bpb = N_ / 256;                 // blocks per batch = 32
    const int b   = blockIdx.x / bpb;
    const int i   = (blockIdx.x % bpb) * 256 + threadIdx.x;
    const float* cbs = centers + (size_t)b * S1 * 3;
    for (int j = threadIdx.x; j < S1; j += 256) {
        sx[j] = cbs[j * 3 + 0];
        sy[j] = cbs[j * 3 + 1];
        sz[j] = cbs[j * 3 + 2];
    }
    __syncthreads();

    const size_t r = (size_t)b * N_ + i;
    const float px = coords[r * 3 + 0];
    const float py = coords[r * 3 + 1];
    const float pz = coords[r * 3 + 2];

    float d0 = 3.4e38f, d1 = 3.4e38f, d2 = 3.4e38f;
    int   i0 = 0, i1 = 0, i2 = 0;
    for (int j = 0; j < S1; ++j) {
        const float dx = sx[j] - px, dy = sy[j] - py, dz = sz[j] - pz;
        const float d  = dx * dx + dy * dy + dz * dz;
        if (d < d0)      { d2 = d1; i2 = i1; d1 = d0; i1 = i0; d0 = d; i0 = j; }
        else if (d < d1) { d2 = d1; i2 = i1; d1 = d;  i1 = j; }
        else if (d < d2) { d2 = d;  i2 = j; }
    }
    float w0 = 1.0f / (d0 + 1e-8f);
    float w1 = 1.0f / (d1 + 1e-8f);
    float w2 = 1.0f / (d2 + 1e-8f);
    const float ws = w0 + w1 + w2;
    w0 /= ws; w1 /= ws; w2 /= ws;
    interp[r * 3 + 0] = w0 * sx[i0] + w1 * sx[i1] + w2 * sx[i2];
    interp[r * 3 + 1] = w0 * sy[i0] + w1 * sy[i1] + w2 * sy[i2];
    interp[r * 3 + 2] = w0 * sz[i0] + w1 * sz[i1] + w2 * sz[i2];
}

// ---------------------------------------------------------------------------
// Kernel 3: WMMA MLP. 128 threads = 4 waves, 16 rows/wave, 64 rows/block.
// A frag (16x32 f16): lane l holds row (l&15); VGPR v -> K = (v>>2)*16 +
// (l>>4)*8 + (v&3)*2 (packed pair). B frag (32x16 f16) from Wt[cout][cin]:
// lane l holds col (l&15); VGPR v -> K = (v>>2)*16 + (l>>4)*8 + (v&3)*2. D: lane l -> col
// (l&15), VGPR v -> row v + 8*(l>>4). K-tiles accumulate through the C operand.
// ---------------------------------------------------------------------------
template <int CINP, bool FINAL>
__device__ __forceinline__ void mlp_layer(
    const _Float16* __restrict__ actIn, const _Float16* __restrict__ wt,
    const float* __restrict__ bias, const float* __restrict__ gamma,
    const float* __restrict__ beta, _Float16* __restrict__ actOut,
    float* __restrict__ gOut, size_t gRowBase, int lane, int waveRow0)
{
    const int half  = lane >> 4;
    const int idx16 = lane & 15;
#pragma unroll
    for (int nt = 0; nt < CH / 16; ++nt) {
        v8f acc = {0.f, 0.f, 0.f, 0.f, 0.f, 0.f, 0.f, 0.f};
#pragma unroll
        for (int kt = 0; kt < CINP / 32; ++kt) {
            FragAB a, bb;
#pragma unroll
            for (int v = 0; v < 8; ++v) {
                const int ka = kt * 32 + (v >> 2) * 16 + half * 8 + (v & 3) * 2;
                a.u[v]  = *(const unsigned*)(actIn + (waveRow0 + idx16) * CINP + ka);
                const int kb = kt * 32 + (v >> 2) * 16 + half * 8 + (v & 3) * 2;
                bb.u[v] = *(const unsigned*)(wt + (nt * 16 + idx16) * CINP + kb);
            }
            acc = __builtin_amdgcn_wmma_f32_16x16x32_f16(
                false, a.v, false, bb.v, (short)0, acc, false, false);
        }
        const int n = nt * 16 + idx16;
        const float s = gamma[n] * BN_INV_F;           // folded BN scale
        const float t = s * bias[n] + beta[n];         // folded BN shift
#pragma unroll
        for (int v = 0; v < 8; ++v) {
            const int mr = waveRow0 + v + 8 * half;
            float y = acc[v] * s + t;
            y = y > 0.0f ? y : 0.0f;
            if (FINAL) gOut[(gRowBase + mr) * CH + n] = y;
            else       actOut[mr * CH + n] = (_Float16)y;
        }
    }
}

__global__ __launch_bounds__(128) void fp1_mlp_kernel(
    const float* __restrict__ data, const float* __restrict__ interp,
    const float* __restrict__ W1, const float* __restrict__ b1,
    const float* __restrict__ g1, const float* __restrict__ be1,
    const float* __restrict__ W2, const float* __restrict__ b2,
    const float* __restrict__ g2, const float* __restrict__ be2,
    const float* __restrict__ W3, const float* __restrict__ b3,
    const float* __restrict__ g3, const float* __restrict__ be3,
    float* __restrict__ out)
{
    __shared__ __align__(16) _Float16 wbuf[CH * CH];   // current layer Wt [cout][cin]
    __shared__ __align__(16) _Float16 actA[64 * CH];
    __shared__ __align__(16) _Float16 actB[64 * CH];   // first 64*32 doubles as L1 input

    const int tid  = threadIdx.x;
    const int lane = tid & 31, wave = tid >> 5;
    const int wr0  = wave * 16;
    const size_t rowBase = (size_t)blockIdx.x * 64;

    // Stage layer-1 input rows: [data(6), interp(3), 0-pad] -> 64 x 32 f16
    for (int idx = tid; idx < 64 * 32; idx += 128) {
        const int row = idx >> 5, c = idx & 31;
        const size_t R = rowBase + row;
        float v = 0.0f;
        if (c < DDIM)          v = data[R * DDIM + c];
        else if (c < DDIM + 3) v = interp[R * 3 + (c - DDIM)];
        actB[idx] = (_Float16)v;
    }
    // Stage W1^T (128 x 32, cin padded 9 -> 32)
    for (int idx = tid; idx < CH * 32; idx += 128) {
        const int co = idx >> 5, ci = idx & 31;
        wbuf[idx] = (_Float16)(ci < 9 ? W1[ci * CH + co] : 0.0f);
    }
    __syncthreads();
    mlp_layer<32, false>(actB, wbuf, b1, g1, be1, actA, nullptr, 0, lane, wr0);
    __syncthreads();

    for (int idx = tid; idx < CH * CH; idx += 128)      // W2^T
        wbuf[idx] = (_Float16)W2[(idx & 127) * CH + (idx >> 7)];
    __syncthreads();
    mlp_layer<CH, false>(actA, wbuf, b2, g2, be2, actB, nullptr, 0, lane, wr0);
    __syncthreads();

    for (int idx = tid; idx < CH * CH; idx += 128)      // W3^T
        wbuf[idx] = (_Float16)W3[(idx & 127) * CH + (idx >> 7)];
    __syncthreads();
    mlp_layer<CH, true>(actB, wbuf, b3, g3, be3, nullptr, out, rowBase, lane, wr0);
}

// ---------------------------------------------------------------------------
// Input leaf order: 0 coords, 1 data, 2..13 sa1, 14..25 sa2, 26..33 fp2,
// 34..45 fp1 (3 layers x [W, bias, gamma, beta]). Only coords/data/fp1 used.
// ---------------------------------------------------------------------------
extern "C" void kernel_launch(void* const* d_in, const int* in_sizes, int n_in,
                              void* d_out, int out_size, void* d_ws, size_t ws_size,
                              hipStream_t stream)
{
    (void)in_sizes; (void)n_in; (void)out_size; (void)ws_size;
    const float* coords = (const float*)d_in[0];
    const float* data   = (const float*)d_in[1];
    const float* W1  = (const float*)d_in[34];
    const float* b1  = (const float*)d_in[35];
    const float* g1  = (const float*)d_in[36];
    const float* be1 = (const float*)d_in[37];
    const float* W2  = (const float*)d_in[38];
    const float* b2  = (const float*)d_in[39];
    const float* g2  = (const float*)d_in[40];
    const float* be2 = (const float*)d_in[41];
    const float* W3  = (const float*)d_in[42];
    const float* b3  = (const float*)d_in[43];
    const float* g3  = (const float*)d_in[44];
    const float* be3 = (const float*)d_in[45];

    float* centers = (float*)d_ws;                       // B*S1*3 floats
    float* interp  = centers + (size_t)B_ * S1 * 3;      // B*N*3 floats

    fps_centers_kernel<<<dim3(B_), dim3(1024), 0, stream>>>(coords, centers);
    knn3_interp_kernel<<<dim3(B_ * (N_ / 256)), dim3(256), 0, stream>>>(coords, centers, interp);
    fp1_mlp_kernel<<<dim3((B_ * N_) / 64), dim3(128), 0, stream>>>(
        data, interp, W1, b1, g1, be1, W2, b2, g2, be2, W3, b3, g3, be3,
        (float*)d_out);
}
